// CrossAttnTimeQueryHead_56143812494028
// MI455X (gfx1250) — matrix operand order repaired and checked by reading, not compiled, not run on hardware
//
#include <hip/hip_runtime.h>
#include <math.h>

// ---------------------------------------------------------------------------
// Types for CDNA5 WMMA (gfx1250, wave32)
// ---------------------------------------------------------------------------
typedef __bf16 bf16_t;
typedef bf16_t v16bf __attribute__((ext_vector_type(16)));
typedef float  v8f   __attribute__((ext_vector_type(8)));
typedef unsigned int u32x4 __attribute__((ext_vector_type(4)));
typedef int          i32x8 __attribute__((ext_vector_type(8)));
typedef int          i32x4 __attribute__((ext_vector_type(4)));

__device__ inline unsigned short f2bfbits(float f) {
    unsigned u; __builtin_memcpy(&u, &f, 4);
    unsigned r = u + 0x7FFFu + ((u >> 16) & 1u);   // round-to-nearest-even
    return (unsigned short)(r >> 16);
}
__device__ inline bf16_t f2bf(float f) {
    unsigned short h = f2bfbits(f);
    bf16_t b; __builtin_memcpy(&b, &h, 2); return b;
}

__device__ inline v8f wmma_bf16(v16bf a, v16bf b, v8f c) {
    return __builtin_amdgcn_wmma_f32_16x16x32_bf16(false, a, false, b, (short)0, c,
                                                   false, false);
}

// A fragment = two contiguous 8-elem bf16 runs: K=[k0,k0+8) and K=[k0+16,k0+24)
__device__ inline void loadAfrag(v16bf& a, const unsigned short* rowBase, int k0) {
    __builtin_memcpy(&a, rowBase + k0, 16);
    __builtin_memcpy(((char*)&a) + 16, rowBase + k0 + 16, 16);
}
__device__ inline void loadAfrag(v16bf& a, const float* rowBase, int k0) {
    float t0[4], t1[4], t2[4], t3[4];
    __builtin_memcpy(t0, rowBase + k0,      16);
    __builtin_memcpy(t1, rowBase + k0 + 4,  16);
    __builtin_memcpy(t2, rowBase + k0 + 16, 16);
    __builtin_memcpy(t3, rowBase + k0 + 20, 16);
    #pragma unroll
    for (int j = 0; j < 4; ++j) {
        a[j]      = f2bf(t0[j]);
        a[j + 4]  = f2bf(t1[j]);
        a[j + 8]  = f2bf(t2[j]);
        a[j + 12] = f2bf(t3[j]);
    }
}
// B fragment = two contiguous 8-elem bf16 runs: K=[k0,k0+8) and K=[k0+16,k0+24)
__device__ inline void loadBfrag(v16bf& b, const unsigned short* src) {
    __builtin_memcpy(&b, src, 16);
    __builtin_memcpy(((char*)&b) + 16, src + 16, 16);
}

// LDS byte offset of a shared-memory pointer (generic addr low 32 bits)
__device__ inline unsigned ldsOffsetOf(const void* p) {
    return (unsigned)(unsigned long long)p;
}

// ---------------------------------------------------------------------------
// WMMA GEMM:  out = act( A[M,K] * W[K,N] + bias + pos + resid )
// Block = 128 threads (4 waves) covering 4 M-tiles x one N-tile.
// W chunk (32x16 f32) staged per block into LDS transposed [n][k] bf16.
// ---------------------------------------------------------------------------
template <typename AT>
__global__ void __launch_bounds__(128)
gemm_wmma_kernel(const AT* __restrict__ A, const float* __restrict__ W,
                 const float* __restrict__ bias,
                 const float* __restrict__ pos, int posMod,
                 const float* __restrict__ resid, int act,
                 float* __restrict__ outF, unsigned short* __restrict__ outB,
                 int M, int K, int N)
{
    const int lane = threadIdx.x & 31;
    const int wave = threadIdx.x >> 5;
    const int nBlocksN = N >> 4;
    const int bm = blockIdx.x / nBlocksN;       // 64-row block
    const int nt = blockIdx.x - bm * nBlocksN;  // 16-col tile
    const int half = lane >> 4;
    const int l16  = lane & 15;
    const int mA   = bm * 64 + wave * 16 + l16;
    const int nB   = nt * 16 + l16;
    const int k0A  = half ? 8  : 0;
    const int k0B  = half ? 8  : 0;

    __shared__ __align__(16) unsigned short WshT[16 * 32];  // [n][k] bf16

    v8f acc;
    #pragma unroll
    for (int v = 0; v < 8; ++v) acc[v] = 0.0f;

    const AT* Arow = A + (long)mA * K;

    for (int kc = 0; kc < K; kc += 32) {
        // stage W chunk transposed (512 elems / 128 threads = 4 each)
        {
            const int i0 = threadIdx.x * 4;               // i = k*16 + n
            float w4[4];
            __builtin_memcpy(w4, &W[(long)(kc + (i0 >> 4)) * N + nt * 16 + (i0 & 15)], 16);
            #pragma unroll
            for (int u = 0; u < 4; ++u) {
                const int i = i0 + u;
                WshT[(i & 15) * 32 + (i >> 4)] = f2bfbits(w4[u]);
            }
        }
        if (kc + 32 < K)
            __builtin_prefetch(&W[(long)(kc + 32) * N + nt * 16 + (threadIdx.x & 15)], 0, 0);
        __syncthreads();

        v16bf a, b;
        loadAfrag(a, Arow + kc, k0A);
        loadBfrag(b, &WshT[l16 * 32 + k0B]);
        acc = wmma_bf16(a, b, acc);
        __syncthreads();
    }

    #pragma unroll
    for (int v = 0; v < 8; ++v) {
        const int m = bm * 64 + wave * 16 + v + 8 * half;
        float val = acc[v];
        if (bias)  val += bias[nB];
        if (pos)   val += pos[(long)(m % posMod) * N + nB];
        if (resid) val += resid[(long)m * N + nB];
        if (act == 1)   // exact GELU
            val = 0.5f * val * (1.0f + erff(val * 0.70710678118654752f));
        const long idx = (long)m * N + nB;
        if (outF) outF[idx] = val;
        if (outB) outB[idx] = f2bfbits(val);
    }
}

// ---------------------------------------------------------------------------
// Flash cross-attention for one (batch, head).
// Q staged via global_load_async_to_lds_b128 (ASYNCcnt path).
// V chunk staged via Tensor Data Mover (TENSORcnt path), then transposed in LDS.
// ---------------------------------------------------------------------------
__global__ void __launch_bounds__(256)
attn_kernel(const unsigned short* __restrict__ Qp,
            const unsigned short* __restrict__ Kp,
            const unsigned short* __restrict__ Vp,
            unsigned short* __restrict__ Obuf,
            int Tin)
{
    const int bh   = blockIdx.x;
    const int b    = bh >> 3;
    const int h    = bh & 7;
    const int tid  = threadIdx.x;
    const int lane = tid & 31;
    const int wave = tid >> 5;
    const int half = lane >> 4;
    const int l16  = lane & 15;

    __shared__ __align__(16) float          Ssh[64 * 64];
    __shared__ __align__(16) unsigned short Psh[64 * 64];
    __shared__ __align__(16) unsigned short Qsh[64 * 32];
    __shared__ __align__(16) unsigned short Vsh[64 * 32];    // [t][e] bf16 (TDM dest)
    __shared__ __align__(16) unsigned short VshT[32 * 64];   // [e][t] bf16
    __shared__ float rowMax[64], rowSum[64], rowScale[64];
    __shared__ float red[64 * 4];

    // ---- stage Q tile asynchronously: 16B per thread, layout-preserving ----
    {
        const int i = tid * 8;                       // u16 elements
        const int r = i >> 5, c = i & 31;
        const unsigned ldsA = ldsOffsetOf(&Qsh[i]);
        const unsigned long long ga =
            (unsigned long long)(const void*)&Qp[((b * 64 + r) * 256) + h * 32 + c];
        asm volatile("global_load_async_to_lds_b128 %0, %1, off"
                     :: "v"(ldsA), "v"(ga) : "memory");
    }
    if (tid < 64) { rowMax[tid] = -3.0e38f; rowSum[tid] = 0.0f; }
    asm volatile("s_wait_asynccnt 0" ::: "memory");
    __syncthreads();

    const int qtO = wave >> 1;      // O q-tile  (0..3)
    const int ntO = wave & 1;       // O head-col tile (0..1)
    v8f oacc;
    #pragma unroll
    for (int v = 0; v < 8; ++v) oacc[v] = 0.0f;

    const float scl = 0.17677669529663687f;   // HEAD^-0.5
    const unsigned ldsV = ldsOffsetOf(&Vsh[0]);

    for (int kc = 0; kc < 16; ++kc) {
        const int kbase = kc * 64;

        // ---- TDM: load V tile [rows=64 x cols=32 bf16] into Vsh ----
        if (wave == 0) {
            const unsigned long long ga = (unsigned long long)(const void*)
                &Vp[((size_t)(b * 1000 + kbase)) * 256 + h * 32];
            const unsigned rowsLeft = (unsigned)(Tin - kbase);  // OOB rows -> 0
            u32x4 g0;
            g0[0] = 1u;                                   // count=1
            g0[1] = ldsV;                                 // lds_addr
            g0[2] = (unsigned)(ga & 0xFFFFFFFFu);         // global_addr lo
            g0[3] = (unsigned)((ga >> 32) & 0x01FFFFFFu)  // global_addr hi
                  | 0x80000000u;                          // type=2 ("image")
            i32x8 g1;
            g1[0] = 0x00010000;                           // data_size=1 (2B)
            g1[1] = (int)((256u & 0xFFFFu) << 16);        // tensor_dim0=256 (lo16)
            g1[2] = (int)((256u >> 16) & 0xFFFFu)
                  | (int)((rowsLeft & 0xFFFFu) << 16);    // dim0 hi | tensor_dim1 lo
            g1[3] = (int)((rowsLeft >> 16) & 0xFFFFu)
                  | (32 << 16);                           // dim1 hi | tile_dim0=32
            g1[4] = 64;                                   // tile_dim1=64, tile_dim2=0
            g1[5] = 256;                                  // tensor_dim0_stride lo32
            g1[6] = 0;
            g1[7] = 0;
            i32x4 gz;  gz[0] = 0;  gz[1] = 0;  gz[2] = 0;  gz[3] = 0;
            i32x8 gz8; gz8[0] = 0; gz8[1] = 0; gz8[2] = 0; gz8[3] = 0;
            gz8[4] = 0; gz8[5] = 0; gz8[6] = 0; gz8[7] = 0;
            __builtin_amdgcn_tensor_load_to_lds(g0, g1, gz, gz, gz8, 0);
        }

        // ---- S = Q * K^T for this 64x64 chunk (16 tiles, 2 per wave) ----
        for (int tt = 0; tt < 2; ++tt) {
            const int t  = wave + tt * 8;
            const int qt = t >> 2, ct = t & 3;
            const int tcol = kbase + ct * 16 + l16;           // time (N dim)
            const int tc   = min(tcol, Tin - 1);
            v16bf a, bb;
            loadAfrag(a, &Qsh[(qt * 16 + l16) * 32], half ? 8 : 0);
            loadBfrag(bb, &Kp[((b * 1000 + tc) * 256) + h * 32 + (half ? 8 : 0)]);
            v8f s;
            #pragma unroll
            for (int v = 0; v < 8; ++v) s[v] = 0.0f;
            s = wmma_bf16(a, bb, s);
            #pragma unroll
            for (int v = 0; v < 8; ++v) {
                const int row = qt * 16 + v + 8 * half;
                const int col = ct * 16 + l16;
                Ssh[row * 64 + col] =
                    (kbase + col < Tin) ? s[v] * scl : -1.0e30f;
            }
        }
        if (wave == 0) __builtin_amdgcn_s_wait_tensorcnt(0);
        __syncthreads();     // Ssh ready + V tile landed in Vsh

        // ---- transpose Vsh [t][e] -> VshT [e][t] (8 elems / thread) ----
        {
            const int i = tid * 8;
            const int tl = i >> 5, e0 = i & 31;
            unsigned short v8[8];
            __builtin_memcpy(v8, &Vsh[tl * 32 + e0], 16);
            #pragma unroll
            for (int j = 0; j < 8; ++j)
                VshT[(e0 + j) * 64 + tl] = v8[j];
        }

        // ---- online softmax: chunk max ----
        {
            const int r = tid >> 2, p = tid & 3;
            float mx = -3.0e38f;
            for (int c = p * 16; c < p * 16 + 16; ++c)
                mx = fmaxf(mx, Ssh[r * 64 + c]);
            red[r * 4 + p] = mx;
        }
        __syncthreads();
        if ((tid & 3) == 0) {
            const int r = tid >> 2;
            const float cm = fmaxf(fmaxf(red[r*4], red[r*4+1]),
                                   fmaxf(red[r*4+2], red[r*4+3]));
            const float mOld = rowMax[r];
            const float mNew = fmaxf(mOld, cm);
            rowScale[r] = __expf(mOld - mNew);
            rowMax[r]   = mNew;
        }
        __syncthreads();

        // ---- P = exp(S - m), partial sums ----
        {
            const int r = tid >> 2, p = tid & 3;
            const float mNew = rowMax[r];
            float s = 0.0f;
            for (int c = p * 16; c < p * 16 + 16; ++c) {
                const float e = __expf(Ssh[r * 64 + c] - mNew);
                Psh[r * 64 + c] = f2bfbits(e);
                s += e;
            }
            red[r * 4 + p] = s;
        }
        __syncthreads();
        if ((tid & 3) == 0) {
            const int r = tid >> 2;
            rowSum[r] = rowSum[r] * rowScale[r]
                      + red[r*4] + red[r*4+1] + red[r*4+2] + red[r*4+3];
        }

        // ---- O = O*scale + P * V  (K dim = 64 -> two x32 WMMAs) ----
        #pragma unroll
        for (int v = 0; v < 8; ++v)
            oacc[v] *= rowScale[qtO * 16 + v + 8 * half];

        #pragma unroll
        for (int kk = 0; kk < 64; kk += 32) {
            v16bf a, bb;
            loadAfrag(a, &Psh[(qtO * 16 + l16) * 64 + kk], half ? 8 : 0);
            loadBfrag(bb, &VshT[(ntO * 16 + l16) * 64 + kk + (half ? 8 : 0)]);
            oacc = wmma_bf16(a, bb, oacc);
        }
        __syncthreads();
    }

    #pragma unroll
    for (int v = 0; v < 8; ++v) {
        const int row = qtO * 16 + v + 8 * half;
        const int col = h * 32 + ntO * 16 + l16;
        const float val = oacc[v] / rowSum[row];
        Obuf[((b * 64 + row) * 256) + col] = f2bfbits(val);
    }
}

// ---------------------------------------------------------------------------
// q <- LayerNorm(q + attn) * s + b ; also write bf16 copy. D = 256.
// ---------------------------------------------------------------------------
__global__ void __launch_bounds__(256)
add_ln_kernel(float* __restrict__ q, const float* __restrict__ attn,
              const float* __restrict__ s, const float* __restrict__ b2,
              unsigned short* __restrict__ qbf)
{
    const int row = blockIdx.x, tid = threadIdx.x;
    __shared__ float red[256];
    const long idx = (long)row * 256 + tid;
    const float x = q[idx] + attn[idx];
    red[tid] = x; __syncthreads();
    for (int st = 128; st > 0; st >>= 1) {
        if (tid < st) red[tid] += red[tid + st];
        __syncthreads();
    }
    const float mu = red[0] * (1.0f / 256.0f);
    __syncthreads();
    const float d = x - mu;
    red[tid] = d * d; __syncthreads();
    for (int st = 128; st > 0; st >>= 1) {
        if (tid < st) red[tid] += red[tid + st];
        __syncthreads();
    }
    const float var = red[0] * (1.0f / 256.0f);
    const float y = d * rsqrtf(var + 1e-5f) * s[tid] + b2[tid];
    q[idx]   = y;
    qbf[idx] = f2bfbits(y);
}

// ---------------------------------------------------------------------------
// Broadcast time_queries[64,256] -> q[B,64,256]
// ---------------------------------------------------------------------------
__global__ void __launch_bounds__(256)
init_q_kernel(const float* __restrict__ tq, float* __restrict__ q, int total)
{
    const int i = blockIdx.x * 256 + threadIdx.x;
    if (i < total) q[i] = tq[i % (64 * 256)];
}

// ---------------------------------------------------------------------------
// Host-side orchestration
// ---------------------------------------------------------------------------
extern "C" void kernel_launch(void* const* d_in, const int* in_sizes, int n_in,
                              void* d_out, int out_size, void* d_ws, size_t ws_size,
                              hipStream_t stream)
{
    (void)in_sizes; (void)n_in; (void)out_size; (void)ws_size;
    const float* x    = (const float*)d_in[0];
    const float* tq   = (const float*)d_in[1];
    const float* pos  = (const float*)d_in[2];
    const float* win  = (const float*)d_in[3];
    const float* bin  = (const float*)d_in[4];
    const float* qw   = (const float*)d_in[5];
    const float* qb   = (const float*)d_in[6];
    const float* kw   = (const float*)d_in[7];
    const float* kb   = (const float*)d_in[8];
    const float* vw   = (const float*)d_in[9];
    const float* vb   = (const float*)d_in[10];
    const float* ow   = (const float*)d_in[11];
    const float* ob   = (const float*)d_in[12];
    const float* lns  = (const float*)d_in[13];
    const float* lnb  = (const float*)d_in[14];
    const float* f1w  = (const float*)d_in[15];
    const float* f1b  = (const float*)d_in[16];
    const float* f2w  = (const float*)d_in[17];
    const float* f2b  = (const float*)d_in[18];
    const float* wout = (const float*)d_in[19];
    const float* bout = (const float*)d_in[20];
    float* out = (float*)d_out;

    const int Bn = 128, Tin = 1000, Din = 512, Tout = 64, D = 256, Dff = 1024,
              Dout = 512, Lnum = 2;
    const int Mh = Bn * Tin;      // 128000
    const int Mq = Bn * Tout;     // 8192

    char* p = (char*)d_ws;
    auto carve = [&](size_t bytes) {
        char* r = p; p += (bytes + 255) & ~(size_t)255; return r;
    };
    unsigned short* h_bf   = (unsigned short*)carve((size_t)Mh * D * 2);
    unsigned short* Kpb    = (unsigned short*)carve((size_t)Mh * D * 2);
    unsigned short* Vpb    = (unsigned short*)carve((size_t)Mh * D * 2);
    unsigned short* Qpb    = (unsigned short*)carve((size_t)Mq * D * 2);
    unsigned short* Ob     = (unsigned short*)carve((size_t)Mq * D * 2);
    float*          attnO  = (float*)         carve((size_t)Mq * D * 4);
    float*          qst    = (float*)         carve((size_t)Mq * D * 4);
    unsigned short* q_bf   = (unsigned short*)carve((size_t)Mq * D * 2);
    unsigned short* ffn_bf = (unsigned short*)carve((size_t)Mq * Dff * 2);

    auto gemmF = [&](const float* A, const float* W, const float* bias,
                     const float* posp, int posMod, const float* resid, int act,
                     float* oF, unsigned short* oB, int M, int K, int N) {
        const int blocks = (M / 64) * (N / 16);
        gemm_wmma_kernel<float><<<blocks, 128, 0, stream>>>(
            A, W, bias, posp, posMod, resid, act, oF, oB, M, K, N);
    };
    auto gemmB = [&](const unsigned short* A, const float* W, const float* bias,
                     const float* resid, int act,
                     float* oF, unsigned short* oB, int M, int K, int N) {
        const int blocks = (M / 64) * (N / 16);
        gemm_wmma_kernel<unsigned short><<<blocks, 128, 0, stream>>>(
            A, W, bias, nullptr, 1, resid, act, oF, oB, M, K, N);
    };

    // h = x @ win + bin + pos_encoding[:Tin]  -> bf16
    gemmF(x, win, bin, pos, Tin, nullptr, 0, nullptr, h_bf, Mh, Din, D);

    // q = broadcast(time_queries)
    init_q_kernel<<<(Mq * D) / 256, 256, 0, stream>>>(tq, qst, Mq * D);

    for (int i = 0; i < Lnum; ++i) {
        const float* qwi = qw + (size_t)i * D * D;
        const float* kwi = kw + (size_t)i * D * D;
        const float* vwi = vw + (size_t)i * D * D;
        const float* owi = ow + (size_t)i * D * D;

        gemmF(qst, qwi, qb + i * D, nullptr, 1, nullptr, 0, nullptr, Qpb, Mq, D, D);
        gemmB(h_bf, kwi, kb + i * D, nullptr, 0, nullptr, Kpb, Mh, D, D);
        gemmB(h_bf, vwi, vb + i * D, nullptr, 0, nullptr, Vpb, Mh, D, D);

        attn_kernel<<<Bn * 8, 256, 0, stream>>>(Qpb, Kpb, Vpb, Ob, Tin);

        gemmB(Ob, owi, ob + i * D, nullptr, 0, attnO, nullptr, Mq, D, D);

        add_ln_kernel<<<Mq, 256, 0, stream>>>(qst, attnO, lns + i * D,
                                              lnb + i * D, q_bf);

        gemmB(q_bf, f1w + (size_t)i * D * Dff, f1b + i * Dff,
              nullptr, 1 /*gelu*/, nullptr, ffn_bf, Mq, D, Dff);
        gemmB(ffn_bf, f2w + (size_t)i * Dff * D, f2b + i * D,
              qst /*residual*/, 0, qst, nullptr, Mq, Dff, D);
    }

    // out = q @ wout + bout
    gemmF(qst, wout, bout, nullptr, 1, nullptr, 0, out, nullptr, Mq, D, Dout);
}
